// PatchCore_68934225101405
// MI455X (gfx1250) — compile-verified, matrix operand order checked by reading, not run on hardware
//
#include <hip/hip_runtime.h>
#include <hip/hip_bf16.h>
#include <math.h>

typedef __attribute__((ext_vector_type(16))) _Float16 v16h;
typedef __attribute__((ext_vector_type(8)))  float    v8f;

// ---------------- problem constants ----------------
#define N_PATCH   676          // 26*26
#define NP_PAD    688          // 43 tiles of 16
#define N_TILES   43
#define DIM       384
#define KCHUNKS   12           // 384 / 32
#define M_LIB     200000
#define LIB_ROWS_PER_BLK 64    // 4 tiles of 16
#define N_GEMM_BLKS 3125       // ceil(200000/64)
#define IMG       224
#define KS        33
#define RADIUS    16
#define SIGMA_F   4.0f

// ---------------- workspace layout (bytes) ----------------
#define OFF_ARM       0u                        // 688*384 f16      = 528384
#define OFF_AFRAG     528384u                   // 688*384 f16      = 528384
#define OFF_PN        1056768u                  // 688 f32          = 2752
#define OFF_BLOCKMIN  1059520u                  // 3125*688 u64     = 17200000
#define OFF_MINVAL    18259520u                 // 676 f32
#define OFF_MINIDX    18262224u                 // 676 i32
#define OFF_SCAL      18264928u                 // 8 i32 : [0]=sidx [1]=mstaridx [2..6]=nn_idx
#define OFF_SSTAR     18264960u                 // 1 f32
#define OFF_D2STAR    18264976u                 // 200000 f32
#define OFF_TMPA      19064976u                 // 50176 f32 (resized map)
#define OFF_TMPB      19265680u                 // 50176 f32 (blurV result)

static __device__ __forceinline__ unsigned short f16bits(float x) {
    _Float16 h = (_Float16)x;
    unsigned short s;
    __builtin_memcpy(&s, &h, 2);
    return s;
}

// ============ K1: L2-normalize patch rows -> f16 row-major + pn ============
__global__ __launch_bounds__(64) void k_normalize(const float* __restrict__ patch,
                                                  _Float16* __restrict__ arm,
                                                  float* __restrict__ pn) {
    int row = blockIdx.x;
    int tid = threadIdx.x;
    __shared__ float red[64];
    if (row >= N_PATCH) {                 // zero padding rows
        for (int k = tid; k < DIM; k += 64) arm[(size_t)row * DIM + k] = (_Float16)0.0f;
        if (tid == 0) pn[row] = 0.0f;
        return;
    }
    float s = 0.0f;
    for (int k = tid; k < DIM; k += 64) {
        float x = patch[(size_t)row * DIM + k];
        s += x * x;
    }
    red[tid] = s;
    __syncthreads();
    for (int st = 32; st > 0; st >>= 1) {
        if (tid < st) red[tid] += red[tid + st];
        __syncthreads();
    }
    float tot = red[0];
    float inv = 1.0f / fmaxf(sqrtf(tot), 1e-12f);
    for (int k = tid; k < DIM; k += 64)
        arm[(size_t)row * DIM + k] = (_Float16)(patch[(size_t)row * DIM + k] * inv);
    if (tid == 0) pn[row] = tot * inv * inv;
}

// ============ K2: pack A into CDNA5 16-bit A-fragment layout ============
// lane l: m = l&15, hi = l>>4 ; dword v holds K pair starting at
//   k0 = kc*32 + 2*(v&3) + 8*hi + 16*(v>>2)
__global__ __launch_bounds__(32) void k_packA(const _Float16* __restrict__ arm,
                                              _Float16* __restrict__ afrag) {
    int a    = blockIdx.x / KCHUNKS;
    int kc   = blockIdx.x % KCHUNKS;
    int lane = threadIdx.x;
    int mm = lane & 15, hi = lane >> 4;
    size_t row = (size_t)(a * 16 + mm) * DIM;
    v16h out;
#pragma unroll
    for (int v = 0; v < 8; ++v) {
        int k0 = kc * 32 + 2 * (v & 3) + 8 * hi + 16 * (v >> 2);
        out[2 * v]     = arm[row + k0];
        out[2 * v + 1] = arm[row + k0 + 1];
    }
    *(v16h*)(afrag + ((size_t)(a * KCHUNKS + kc) * 32 + lane) * 16) = out;
}

// ============ K3: WMMA GEMM + per-row running min ============
__global__ __launch_bounds__(256) void k_gemm_min(const float* __restrict__ lib,
                                                  const _Float16* __restrict__ afragG,
                                                  const float* __restrict__ pn,
                                                  unsigned long long* __restrict__ blockmin) {
    // 4 lib tiles x 12 kc x 32 lanes x 8 dwords = 48KB of pre-swizzled f16 B fragments
    __shared__ __align__(16) unsigned int Bfrag[4 * KCHUNKS * 32 * 8];
    __shared__ float lnS[LIB_ROWS_PER_BLK];

    int tid = threadIdx.x;
    long libBase = (long)blockIdx.x * LIB_ROWS_PER_BLK;

    if (tid < LIB_ROWS_PER_BLK) lnS[tid] = 0.0f;
    __syncthreads();

    // stage: convert fp32 lib rows -> f16 B fragments, accumulate ||b||^2
    for (int idx = tid; idx < LIB_ROWS_PER_BLK * (DIM / 2); idx += 256) {
        int r  = idx / (DIM / 2);
        int k  = (idx % (DIM / 2)) * 2;
        long gRow = libBase + r;
        float x0 = 0.0f, x1 = 0.0f;
        if (gRow < M_LIB) {
            const float* p = lib + (size_t)gRow * DIM + k;
            x0 = p[0];
            x1 = p[1];
        }
        atomicAdd(&lnS[r], x0 * x0 + x1 * x1);
        int kc = k >> 5, kk = k & 31;
        int hi = (kk >> 3) & 1;
        int v  = ((kk & 16) >> 2) + ((kk & 7) >> 1);
        int bl = (r & 15) + (hi << 4);
        int t  = r >> 4;
        unsigned u = ((unsigned)f16bits(x1) << 16) | (unsigned)f16bits(x0);
        Bfrag[(((t * KCHUNKS + kc) * 32 + bl) << 3) + v] = u;
    }
    __syncthreads();
    if (tid < LIB_ROWS_PER_BLK && (libBase + tid) >= M_LIB) lnS[tid] = INFINITY;
    __syncthreads();

    int wave = tid >> 5, lane = tid & 31;
    int m = lane & 15, hi = lane >> 4;
    const v16h* afragV = (const v16h*)afragG;

    for (int a = wave; a < N_TILES; a += 8) {
        v16h aAll[KCHUNKS];
#pragma unroll
        for (int kc = 0; kc < KCHUNKS; ++kc)
            aAll[kc] = afragV[(size_t)(a * KCHUNKS + kc) * 32 + lane];
        float pnv[8];
#pragma unroll
        for (int v = 0; v < 8; ++v) pnv[v] = pn[a * 16 + v + 8 * hi];

        unsigned long long mk[8];
#pragma unroll
        for (int v = 0; v < 8; ++v) mk[v] = ~0ull;

        for (int t = 0; t < 4; ++t) {
            v8f acc = {};
#pragma unroll
            for (int kc = 0; kc < KCHUNKS; ++kc) {
                v16h b = *(const v16h*)&Bfrag[((t * KCHUNKS + kc) * 32 + lane) << 3];
                acc = __builtin_amdgcn_wmma_f32_16x16x32_f16(
                    false, aAll[kc], false, b, (short)0, acc, false, false);
            }
            float lnv = lnS[t * 16 + m];
            unsigned long long cidx = (unsigned long long)(libBase + t * 16 + m);
#pragma unroll
            for (int v = 0; v < 8; ++v) {
                float d2 = fmaxf(pnv[v] + lnv - 2.0f * acc[v], 0.0f);
                unsigned long long key =
                    ((unsigned long long)__float_as_uint(d2) << 32) | cidx;
                mk[v] = key < mk[v] ? key : mk[v];
            }
        }
        // min across the 16 lanes that share a patch row
#pragma unroll
        for (int v = 0; v < 8; ++v) {
            unsigned long long k = mk[v];
            for (int off = 1; off < 16; off <<= 1) {
                unsigned long long o = __shfl_xor(k, off, 32);
                k = o < k ? o : k;
            }
            if (m == 0)
                blockmin[(size_t)blockIdx.x * NP_PAD + a * 16 + v + 8 * hi] = k;
        }
    }
}

// ============ K4: reduce block mins -> min_val / min_idx per patch row ============
__global__ __launch_bounds__(256) void k_rowmin(const unsigned long long* __restrict__ blockmin,
                                                float* __restrict__ minval,
                                                int* __restrict__ minidx) {
    int row = blockIdx.x;
    int tid = threadIdx.x;
    __shared__ unsigned long long red[256];
    unsigned long long k = ~0ull;
    for (int b = tid; b < N_GEMM_BLKS; b += 256) {
        unsigned long long o = blockmin[(size_t)b * NP_PAD + row];
        k = o < k ? o : k;
    }
    red[tid] = k;
    __syncthreads();
    for (int s = 128; s > 0; s >>= 1) {
        if (tid < s) { unsigned long long o = red[tid + s]; if (o < red[tid]) red[tid] = o; }
        __syncthreads();
    }
    if (tid == 0) {
        unsigned long long r = red[0];
        minval[row] = sqrtf(__uint_as_float((unsigned)(r >> 32)));
        minidx[row] = (int)(unsigned)(r & 0xFFFFFFFFull);
    }
}

// ============ K5: argmax over min_val -> s_idx, s_star, m_star idx ============
__global__ __launch_bounds__(256) void k_argmax(const float* __restrict__ minval,
                                                const int* __restrict__ minidx,
                                                int* __restrict__ scal,
                                                float* __restrict__ sstar) {
    int tid = threadIdx.x;
    __shared__ unsigned long long red[256];
    unsigned long long k = 0;
    for (int i = tid; i < N_PATCH; i += 256) {
        unsigned long long key =
            ((unsigned long long)__float_as_uint(minval[i]) << 32) |
            (unsigned long long)(0xFFFFFFFFu - (unsigned)i);
        k = key > k ? key : k;
    }
    red[tid] = k;
    __syncthreads();
    for (int s = 128; s > 0; s >>= 1) {
        if (tid < s) { unsigned long long o = red[tid + s]; if (o > red[tid]) red[tid] = o; }
        __syncthreads();
    }
    if (tid == 0) {
        int sidx = (int)(0xFFFFFFFFu - (unsigned)(red[0] & 0xFFFFFFFFull));
        scal[0] = sidx;
        scal[1] = minidx[sidx];
        sstar[0] = minval[sidx];
    }
}

// ============ K6: d_star^2[m] = ||lib[m]||^2 + ||m*||^2 - 2 lib[m].m* ============
__global__ __launch_bounds__(256) void k_matvec(const float* __restrict__ lib,
                                                const int* __restrict__ scal,
                                                float* __restrict__ d2star) {
    __shared__ float ms[DIM];
    int msi = scal[1];
    for (int t = threadIdx.x; t < DIM; t += 256) ms[t] = lib[(size_t)msi * DIM + t];
    __syncthreads();
    int wave = threadIdx.x >> 5, lane = threadIdx.x & 31;
    long row = (long)blockIdx.x * 8 + wave;
    if (row >= M_LIB) return;
    float dot = 0.0f, ln = 0.0f, mn = 0.0f;
    for (int k = lane; k < DIM; k += 32) {
        float a = lib[(size_t)row * DIM + k];
        float b = ms[k];
        dot += a * b; ln += a * a; mn += b * b;
    }
    for (int off = 1; off < 32; off <<= 1) {
        dot += __shfl_xor(dot, off, 32);
        ln  += __shfl_xor(ln,  off, 32);
        mn  += __shfl_xor(mn,  off, 32);
    }
    if (lane == 0) d2star[row] = fmaxf(ln + mn - 2.0f * dot, 0.0f);
}

// ============ K7: exact top-5 smallest of d_star^2 (index-tie-broken) ============
__global__ __launch_bounds__(1024) void k_top5(const float* __restrict__ d2star,
                                               int* __restrict__ scal) {
    __shared__ unsigned long long L[1024 * 5];
    int tid = threadIdx.x;
    unsigned long long top[5] = {~0ull, ~0ull, ~0ull, ~0ull, ~0ull};
    for (int m = tid; m < M_LIB; m += 1024) {
        unsigned long long key =
            ((unsigned long long)__float_as_uint(d2star[m]) << 32) | (unsigned)m;
        if (key < top[4]) {
            top[4] = key;
#pragma unroll
            for (int j = 4; j > 0; --j)
                if (top[j] < top[j - 1]) {
                    unsigned long long t = top[j]; top[j] = top[j - 1]; top[j - 1] = t;
                }
        }
    }
#pragma unroll
    for (int j = 0; j < 5; ++j) L[tid * 5 + j] = top[j];
    __syncthreads();
    for (int s = 512; s >= 1; s >>= 1) {
        if (tid < s) {
            unsigned long long* A = &L[tid * 5];
            unsigned long long* B = &L[(tid + s) * 5];
            unsigned long long tmp[5];
            int ia = 0, ib = 0;
#pragma unroll
            for (int j = 0; j < 5; ++j)
                tmp[j] = (ib >= 5 || (ia < 5 && A[ia] <= B[ib])) ? A[ia++] : B[ib++];
#pragma unroll
            for (int j = 0; j < 5; ++j) A[j] = tmp[j];
        }
        __syncthreads();
    }
    if (tid == 0)
        for (int j = 0; j < 5; ++j)
            scal[2 + j] = (int)(unsigned)(L[j] & 0xFFFFFFFFull);
}

// ============ K8: final scalar score (single wave, fp32) ============
__global__ __launch_bounds__(32) void k_score(const float* __restrict__ patch,
                                              const float* __restrict__ lib,
                                              const int* __restrict__ scal,
                                              const float* __restrict__ sstar,
                                              float* __restrict__ out) {
    int lane = threadIdx.x;
    int sidx = scal[0], msi = scal[1];
    // normalize patch[sidx]
    float s2 = 0.0f;
    for (int k = lane; k < DIM; k += 32) {
        float x = patch[(size_t)sidx * DIM + k];
        s2 += x * x;
    }
    for (int off = 1; off < 32; off <<= 1) s2 += __shfl_xor(s2, off, 32);
    float inv = 1.0f / fmaxf(sqrtf(s2), 1e-12f);

    float dms = 0.0f, dnb[5] = {0, 0, 0, 0, 0};
    for (int k = lane; k < DIM; k += 32) {
        float mts = patch[(size_t)sidx * DIM + k] * inv;
        float d = mts - lib[(size_t)msi * DIM + k];
        dms += d * d;
#pragma unroll
        for (int j = 0; j < 5; ++j) {
            float dd = mts - lib[(size_t)scal[2 + j] * DIM + k];
            dnb[j] += dd * dd;
        }
    }
    for (int off = 1; off < 32; off <<= 1) {
        dms += __shfl_xor(dms, off, 32);
#pragma unroll
        for (int j = 0; j < 5; ++j) dnb[j] += __shfl_xor(dnb[j], off, 32);
    }
    if (lane == 0) {
        float num = expf(sqrtf(dms));
        float den = 0.0f;
#pragma unroll
        for (int j = 0; j < 5; ++j) den += expf(sqrtf(dnb[j]));
        out[0] = (1.0f - num / den) * sstar[0];
    }
}

// ============ K9: bilinear resize 26x26 -> 224x224 (half-pixel, edge clamp) ============
__global__ __launch_bounds__(256) void k_resize(const float* __restrict__ minval,
                                                float* __restrict__ outA) {
    int idx = blockIdx.x * 256 + threadIdx.x;
    if (idx >= IMG * IMG) return;
    int oy = idx / IMG, ox = idx % IMG;
    const float sc = 26.0f / 224.0f;
    float fy = (oy + 0.5f) * sc - 0.5f;
    float fx = (ox + 0.5f) * sc - 0.5f;
    int y0 = (int)floorf(fy), x0 = (int)floorf(fx);
    float ty = fy - y0, tx = fx - x0;
    int y0c = min(max(y0, 0), 25), y1c = min(max(y0 + 1, 0), 25);
    int x0c = min(max(x0, 0), 25), x1c = min(max(x0 + 1, 0), 25);
    float v00 = minval[y0c * 26 + x0c], v01 = minval[y0c * 26 + x1c];
    float v10 = minval[y1c * 26 + x0c], v11 = minval[y1c * 26 + x1c];
    outA[idx] = (1.0f - ty) * ((1.0f - tx) * v00 + tx * v01) +
                ty * ((1.0f - tx) * v10 + tx * v11);
}

// ============ K10/K11: separable Gaussian blur, reflect padding ============
static __device__ __forceinline__ int reflect_idx(int i) {
    if (i < 0) i = -i;
    if (i > IMG - 1) i = 2 * (IMG - 1) - i;
    return i;
}

__global__ __launch_bounds__(256) void k_blur(const float* __restrict__ in,
                                              float* __restrict__ out,
                                              int vertical) {
    __shared__ float w[KS];
    __shared__ float wsum;
    int tid = threadIdx.x;
    if (tid < KS) {
        float x = (float)(tid - RADIUS) / SIGMA_F;
        w[tid] = expf(-0.5f * x * x);
    }
    __syncthreads();
    if (tid == 0) {
        float s = 0.0f;
        for (int j = 0; j < KS; ++j) s += w[j];
        wsum = s;
    }
    __syncthreads();
    int idx = blockIdx.x * 256 + tid;
    if (idx >= IMG * IMG) return;
    int y = idx / IMG, x = idx % IMG;
    float acc = 0.0f;
    if (vertical) {
        for (int j = 0; j < KS; ++j)
            acc += w[j] * in[reflect_idx(y - RADIUS + j) * IMG + x];
    } else {
        for (int j = 0; j < KS; ++j)
            acc += w[j] * in[y * IMG + reflect_idx(x - RADIUS + j)];
    }
    out[idx] = acc / wsum;
}

// ---------------- launcher ----------------
extern "C" void kernel_launch(void* const* d_in, const int* in_sizes, int n_in,
                              void* d_out, int out_size, void* d_ws, size_t ws_size,
                              hipStream_t stream) {
    const float* patch = (const float*)d_in[0];   // [676*384]
    const float* lib   = (const float*)d_in[1];   // [200000*384]
    float* out = (float*)d_out;                   // [1 + 224*224]: s, s_map

    char* ws = (char*)d_ws;
    _Float16* arm   = (_Float16*)(ws + OFF_ARM);
    _Float16* afrag = (_Float16*)(ws + OFF_AFRAG);
    float* pn       = (float*)(ws + OFF_PN);
    unsigned long long* blockmin = (unsigned long long*)(ws + OFF_BLOCKMIN);
    float* minval   = (float*)(ws + OFF_MINVAL);
    int*   minidx   = (int*)(ws + OFF_MINIDX);
    int*   scal     = (int*)(ws + OFF_SCAL);
    float* sstar    = (float*)(ws + OFF_SSTAR);
    float* d2star   = (float*)(ws + OFF_D2STAR);
    float* tmpA     = (float*)(ws + OFF_TMPA);
    float* tmpB     = (float*)(ws + OFF_TMPB);

    k_normalize<<<NP_PAD, 64, 0, stream>>>(patch, arm, pn);
    k_packA<<<N_TILES * KCHUNKS, 32, 0, stream>>>(arm, afrag);
    k_gemm_min<<<N_GEMM_BLKS, 256, 0, stream>>>(lib, afrag, pn, blockmin);
    k_rowmin<<<N_PATCH, 256, 0, stream>>>(blockmin, minval, minidx);
    k_argmax<<<1, 256, 0, stream>>>(minval, minidx, scal, sstar);
    k_matvec<<<(M_LIB + 7) / 8, 256, 0, stream>>>(lib, scal, d2star);
    k_top5<<<1, 1024, 0, stream>>>(d2star, scal);
    k_score<<<1, 32, 0, stream>>>(patch, lib, scal, sstar, out);
    k_resize<<<(IMG * IMG + 255) / 256, 256, 0, stream>>>(minval, tmpA);
    k_blur<<<(IMG * IMG + 255) / 256, 256, 0, stream>>>(tmpA, tmpB, 1);
    k_blur<<<(IMG * IMG + 255) / 256, 256, 0, stream>>>(tmpB, out + 1, 0);
}
